// SparseCoder_63840393888177
// MI455X (gfx1250) — compile-verified, matrix operand order checked by reading, not run on hardware
//
#include <hip/hip_runtime.h>
#include <hip/hip_bf16.h>
#include <stdint.h>

typedef __attribute__((ext_vector_type(16))) __bf16 v16bf;
typedef __attribute__((ext_vector_type(8)))  float  v8f;
typedef __attribute__((ext_vector_type(4)))  unsigned int u32x4;
typedef __attribute__((ext_vector_type(8)))  int  i32x8;
typedef __attribute__((ext_vector_type(4)))  int  i32x4;

#define B_ROWS 4096
#define D_IN   2048
#define NLAT   32768
#define TOPK   32
#define NSLICE 16
#define LSLICE (NLAT / NSLICE)   /* 2048 */
#define MTILE  64
#define NCHUNK 64
#define KSUP   128

/* ---- LDS layout for the fused encode+topk kernel (bytes) ---- */
#define A_STRIDE_B 4112                       /* 2056 bf16: pad kills bank conflicts */
#define A_BYTES    (MTILE * A_STRIDE_B)       /* 263168 */
#define B_STRIDE_B 272                        /* 128 bf16 + 16B TDM pad; 68 words %64 = 4 */
#define BBUF_BYTES (NCHUNK * B_STRIDE_B)      /* 17408 */
#define LDS_B0_OFF   A_BYTES
#define LDS_B1_OFF   (LDS_B0_OFF + BBUF_BYTES)
#define LDS_TV_OFF   (LDS_B1_OFF + BBUF_BYTES)
#define LDS_TI_OFF   (LDS_TV_OFF + MTILE*TOPK*4)
#define LDS_RMIN_OFF (LDS_TI_OFF + MTILE*TOPK*4)
#define LDS_RPOS_OFF (LDS_RMIN_OFF + MTILE*4)
#define LDS_TOTAL    (LDS_RPOS_OFF + MTILE*4) /* 314880 <= 320KB */

/* ---- workspace layout (bytes) ---- */
#define WENC_BYTES ((size_t)NLAT * D_IN * 2)            /* 134217728 */
#define XBF_OFF    (WENC_BYTES)                         /* +16777216 */
#define PVAL_OFF   (XBF_OFF + (size_t)B_ROWS * D_IN * 2)
#define PIDX_OFF   (PVAL_OFF + (size_t)B_ROWS * NSLICE * TOPK * 4)
#define STATS_OFF  (PIDX_OFF + (size_t)B_ROWS * NSLICE * TOPK * 4)

/* ---- output layout (floats) ---- */
#define OUT_ACTS_OFF ((size_t)B_ROWS * D_IN)            /* 8388608 */
#define OUT_IDX_OFF  (OUT_ACTS_OFF + (size_t)B_ROWS * TOPK)
#define OUT_FVU_OFF  (OUT_IDX_OFF + (size_t)B_ROWS * TOPK)

static __device__ __forceinline__ unsigned short f2bf(float f) {
  unsigned int u = __float_as_uint(f);
  u += 0x7fffu + ((u >> 16) & 1u);   /* round to nearest even */
  return (unsigned short)(u >> 16);
}

/* Issue one TDM load: 64 rows x 128 bf16 tile of W_enc -> LDS, with
   16B pad appended every 256B row so LDS row stride = 272B. */
static __device__ __forceinline__ void tdm_load_b(const unsigned short* gsrc,
                                                  unsigned int lds_off) {
  unsigned long long ga = (unsigned long long)(uintptr_t)gsrc;
  u32x4 g0;
  g0[0] = 1u;                                  /* count=1, user mode */
  g0[1] = lds_off;                             /* lds_addr */
  g0[2] = (unsigned int)(ga & 0xffffffffu);    /* global_addr[31:0] */
  g0[3] = (unsigned int)((ga >> 32) & 0x01ffffffu) | (2u << 30); /* [56:32] | type=2 */
  i32x8 g1;
  g1[0] = (int)((1u << 16) |                   /* data_size = 2B */
                (1u << 20) |                   /* pad_enable */
                (5u << 22) |                   /* pad_interval: 64 DWORDs = 256B */
                (3u << 25));                   /* pad_amount: 4 DWORDs = 16B */
  g1[1] = (int)((2048u & 0xffffu) << 16);      /* tensor_dim0 lo16 (dim0=2048) */
  g1[2] = (int)(((2048u >> 16) & 0xffffu) | ((32768u & 0xffffu) << 16)); /* dim0 hi, dim1 lo */
  g1[3] = (int)(((32768u >> 16) & 0xffffu) | ((unsigned)KSUP << 16));    /* dim1 hi, tile_dim0 */
  g1[4] = (int)(64u);                          /* tile_dim1=64, tile_dim2=0 */
  g1[5] = (int)(2048);                         /* tensor_dim0_stride lo32 */
  g1[6] = 0;                                   /* dim0_stride hi, dim1_stride lo */
  g1[7] = 0;
  i32x4 z4 = {0, 0, 0, 0};
  i32x8 z8 = {0, 0, 0, 0, 0, 0, 0, 0};
  __builtin_amdgcn_tensor_load_to_lds(g0, g1, z4, z4, z8, 0);
}

/* ================= stats: column sums for total_variance ================= */
__global__ void kzero(float* stats) {
  int i = blockIdx.x * 256 + threadIdx.x;
  if (i < 4098) stats[i] = 0.0f;
}

__global__ __launch_bounds__(256)
void kcolstats(const float* __restrict__ x, float* __restrict__ colsum,
               float* __restrict__ colsq) {
  const int tid = threadIdx.x;
  const int rb = blockIdx.x * 128;
  float s[8] = {0,0,0,0,0,0,0,0}, q[8] = {0,0,0,0,0,0,0,0};
  for (int r = 0; r < 128; ++r) {
    const float* xr = x + (size_t)(rb + r) * D_IN;
#pragma unroll
    for (int i = 0; i < 8; ++i) { float v = xr[tid + 256*i]; s[i] += v; q[i] += v*v; }
  }
#pragma unroll
  for (int i = 0; i < 8; ++i) {
    atomicAdd(&colsum[tid + 256*i], s[i]);
    atomicAdd(&colsq[tid + 256*i], q[i]);
  }
}

__global__ __launch_bounds__(256)
void ktv(const float* __restrict__ colsum, const float* __restrict__ colsq,
         float* __restrict__ tv) {
  __shared__ float red[256];
  const int tid = threadIdx.x;
  float p = 0.0f;
  for (int i = 0; i < 8; ++i) {
    int c = tid + 256*i;
    float s = colsum[c];
    p += colsq[c] - s * s * (1.0f / (float)B_ROWS);
  }
  red[tid] = p; __syncthreads();
  for (int off = 128; off > 0; off >>= 1) {
    if (tid < off) red[tid] += red[tid + off];
    __syncthreads();
  }
  if (tid == 0) *tv = red[0];
}

/* ================= f32 -> bf16 conversions ================= */
__global__ void kcvt(const float* __restrict__ src, unsigned short* __restrict__ dst, int n4) {
  int idx = blockIdx.x * blockDim.x + threadIdx.x;
  int stride = gridDim.x * blockDim.x;
  for (int i = idx; i < n4; i += stride) {
    float4 v = ((const float4*)src)[i];
    ushort4 o;
    o.x = f2bf(v.x); o.y = f2bf(v.y); o.z = f2bf(v.z); o.w = f2bf(v.w);
    ((ushort4*)dst)[i] = o;
  }
}

__global__ void kcvt_x(const float* __restrict__ x, const float* __restrict__ b_dec,
                       unsigned short* __restrict__ dst, int n4) {
  int idx = blockIdx.x * blockDim.x + threadIdx.x;
  int stride = gridDim.x * blockDim.x;
  for (int i = idx; i < n4; i += stride) {
    float4 v = ((const float4*)x)[i];
    int cb = (i * 4) & (D_IN - 1);
    float4 b = *(const float4*)(b_dec + cb);
    ushort4 o;
    o.x = f2bf(v.x - b.x); o.y = f2bf(v.y - b.y);
    o.z = f2bf(v.z - b.z); o.w = f2bf(v.w - b.w);
    ((ushort4*)dst)[i] = o;
  }
}

/* == fused encode GEMM (bf16 WMMA, TDM-fed, double-buffered) + ReLU + top-32 == */
__global__ __launch_bounds__(256)
void kenc_topk(const unsigned short* __restrict__ Wb,   /* [NLAT][D_IN] bf16 */
               const unsigned short* __restrict__ Xb,   /* [B][D_IN] bf16 = x - b_dec */
               const float* __restrict__ b_enc,
               float* __restrict__ pval, int* __restrict__ pidx)
{
  extern __shared__ char smem[];
  char*  sA   = smem;
  float* topv = (float*)(smem + LDS_TV_OFF);
  int*   topi = (int*)(smem + LDS_TI_OFF);
  float* rmin = (float*)(smem + LDS_RMIN_OFF);
  int*   rpos = (int*)(smem + LDS_RPOS_OFF);

  const int tid = threadIdx.x;
  const int rowBase = blockIdx.x * MTILE;
  const int slice = blockIdx.y;
  const int latSliceBase = slice * LSLICE;

  /* ---- stage the full-K A tile (64 x 2048 bf16) into LDS, padded ---- */
  {
    const uint4* src = (const uint4*)(Xb + (size_t)rowBase * D_IN);
#pragma unroll 4
    for (int i = 0; i < 64; ++i) {
      int j = tid + 256 * i;
      int r = j >> 8;
      int cu = j & 255;
      uint4 v = src[r * 256 + cu];
      *(uint4*)(sA + r * A_STRIDE_B + cu * 16) = v;
    }
  }
  if (tid < MTILE) {
    for (int k = 0; k < TOPK; ++k) { topv[tid*TOPK + k] = -1e30f; topi[tid*TOPK + k] = 0; }
    rmin[tid] = -1e30f; rpos[tid] = 0;
  }
  __syncthreads();

  const int wid = tid >> 5;
  const int lane = tid & 31;
  const int mw = wid & 3;       /* M tile 0..3            */
  const int ng = wid >> 2;      /* N group 0..1 (2 tiles) */
  const int lr = lane & 15;
  const int lh = lane >> 4;

  for (int nc = 0; nc < LSLICE / NCHUNK; ++nc) {
    const int latChunk = latSliceBase + nc * NCHUNK;
    const unsigned short* wrow = Wb + (size_t)latChunk * D_IN;
    v8f acc0 = {}; v8f acc1 = {};

    /* prime the pipeline: superstep 0 -> buffer 0 */
    if (tid < 32) tdm_load_b(wrow, LDS_B0_OFF);

    for (int ks = 0; ks < D_IN / KSUP; ++ks) {
      if (tid < 32) {
        if (ks < D_IN / KSUP - 1) {
          /* prefetch next superstep into the other buffer, then wait
             until the current buffer's DMA (issued earlier) is done */
          tdm_load_b(wrow + (ks + 1) * KSUP,
                     (ks & 1) ? LDS_B0_OFF : LDS_B1_OFF);
          __builtin_amdgcn_s_wait_tensorcnt(1);
        } else {
          __builtin_amdgcn_s_wait_tensorcnt(0);
        }
      }
      __syncthreads();

      char* sB = smem + ((ks & 1) ? LDS_B1_OFF : LDS_B0_OFF);
#pragma unroll
      for (int kk = 0; kk < KSUP / 32; ++kk) {
        const int kb = ks * KSUP + kk * 32;   /* A global-k */
        const int kl = kk * 32;               /* B local-k  */
        v16bf af;
        {
          const __bf16* p0 = (const __bf16*)(sA + (mw*16 + lr) * A_STRIDE_B + (kb + lh*8) * 2);
          const __bf16* p1 = (const __bf16*)(sA + (mw*16 + lr) * A_STRIDE_B + (kb + 16 + lh*8) * 2);
#pragma unroll
          for (int e = 0; e < 8; ++e) { af[e] = p0[e]; af[8+e] = p1[e]; }
        }
        v16bf bf0, bf1;
        {
          const __bf16* q0 = (const __bf16*)(sB + (ng*32 + lr) * B_STRIDE_B + (kl + lh*8) * 2);
          const __bf16* q1 = (const __bf16*)(sB + (ng*32 + lr) * B_STRIDE_B + (kl + 16 + lh*8) * 2);
#pragma unroll
          for (int e = 0; e < 8; ++e) { bf0[e] = q0[e]; bf0[8+e] = q1[e]; }
          const __bf16* r0 = (const __bf16*)(sB + (ng*32 + 16 + lr) * B_STRIDE_B + (kl + lh*8) * 2);
          const __bf16* r1 = (const __bf16*)(sB + (ng*32 + 16 + lr) * B_STRIDE_B + (kl + 16 + lh*8) * 2);
#pragma unroll
          for (int e = 0; e < 8; ++e) { bf1[e] = r0[e]; bf1[8+e] = r1[e]; }
        }
        acc0 = __builtin_amdgcn_wmma_f32_16x16x32_bf16(false, af, false, bf0,
                                                       (short)0, acc0, false, false);
        acc1 = __builtin_amdgcn_wmma_f32_16x16x32_bf16(false, af, false, bf1,
                                                       (short)0, acc1, false, false);
      }
      __syncthreads();   /* all reads of this buffer done before its re-fill */
    }

    /* ---- epilogue: +b_enc, ReLU, scores into LDS (reuse B0), topk update ---- */
    float* sc = (float*)(smem + LDS_B0_OFF);   /* [MTILE][65] floats, 16640B <= 17408 */
    {
      int n0 = ng*32 + lr;
      int n1 = ng*32 + 16 + lr;
      float be0 = b_enc[latChunk + n0];
      float be1 = b_enc[latChunk + n1];
#pragma unroll
      for (int i = 0; i < 8; ++i) {
        int m = mw*16 + lh*8 + i;   /* C layout: vgpr i -> M=i (lanes 0-15), M=8+i (16-31) */
        sc[m*65 + n0] = fmaxf(acc0[i] + be0, 0.0f);
        sc[m*65 + n1] = fmaxf(acc1[i] + be1, 0.0f);
      }
    }
    __syncthreads();
    if (tid < MTILE) {
      const int m = tid;
      float rm = rmin[m]; int rp = rpos[m];
      float* tv = topv + m*TOPK; int* ti = topi + m*TOPK;
      for (int n = 0; n < NCHUNK; ++n) {
        float v = sc[m*65 + n];
        if (v > rm) {
          tv[rp] = v; ti[rp] = latChunk + n;
          rm = tv[0]; rp = 0;
          for (int j = 1; j < TOPK; ++j) { float t = tv[j]; if (t < rm) { rm = t; rp = j; } }
        }
      }
      rmin[m] = rm; rpos[m] = rp;
    }
    __syncthreads();   /* topk readers done before next chunk's TDM refills B0 */
  }

  if (tid < MTILE) {
    int row = rowBase + tid;
    size_t base = ((size_t)row * NSLICE + slice) * TOPK;
    for (int k = 0; k < TOPK; ++k) {
      pval[base + k] = topv[tid*TOPK + k];
      pidx[base + k] = topi[tid*TOPK + k];
    }
  }
}

/* ========== merge 16 partial top-32 lists, sparse decode, error sum ========== */
__global__ __launch_bounds__(256)
void kmerge_decode(const float* __restrict__ x,
                   const float* __restrict__ W_dec,
                   const float* __restrict__ b_dec,
                   const float* __restrict__ pval, const int* __restrict__ pidx,
                   float* __restrict__ out, float* __restrict__ errsum)
{
  __shared__ float cv[512];
  __shared__ int   ci[512];
  __shared__ float redv[256];
  __shared__ int   redp[256];
  __shared__ float stv[TOPK];
  __shared__ int   sti[TOPK];

  const int row = blockIdx.x;
  const int tid = threadIdx.x;
  const size_t pbase = (size_t)row * (NSLICE * TOPK);
  cv[tid]       = pval[pbase + tid];       ci[tid]       = pidx[pbase + tid];
  cv[tid + 256] = pval[pbase + tid + 256]; ci[tid + 256] = pidx[pbase + tid + 256];
  __syncthreads();

  /* 32 rounds of block argmax -> descending-sorted top-32 (matches jax top_k) */
  for (int k = 0; k < TOPK; ++k) {
    float v0 = cv[tid], v1 = cv[tid + 256];
    float bv = v0; int bp = tid;
    if (v1 > bv) { bv = v1; bp = tid + 256; }
    redv[tid] = bv; redp[tid] = bp;
    __syncthreads();
    for (int off = 128; off > 0; off >>= 1) {
      if (tid < off && redv[tid + off] > redv[tid]) {
        redv[tid] = redv[tid + off]; redp[tid] = redp[tid + off];
      }
      __syncthreads();
    }
    if (tid == 0) {
      int p = redp[0];
      stv[k] = redv[0]; sti[k] = ci[p];
      cv[p] = -1e38f;
    }
    __syncthreads();
  }

  if (tid < TOPK) {
    out[OUT_ACTS_OFF + (size_t)row * TOPK + tid] = stv[tid];
    out[OUT_IDX_OFF  + (size_t)row * TOPK + tid] = (float)sti[tid];
  }

  /* decode: each thread owns dims tid + 256*i */
  float acc[8];
#pragma unroll
  for (int i = 0; i < 8; ++i) acc[i] = b_dec[tid + 256*i];
  for (int k = 0; k < TOPK; ++k) {
    float w = stv[k];
    const float* wr = W_dec + (size_t)sti[k] * D_IN;
#pragma unroll
    for (int i = 0; i < 8; ++i) acc[i] += w * wr[tid + 256*i];
  }
  float es = 0.0f;
#pragma unroll
  for (int i = 0; i < 8; ++i) {
    int d = tid + 256*i;
    float o = acc[i];
    out[(size_t)row * D_IN + d] = o;
    float e = x[(size_t)row * D_IN + d] - o;
    es += e * e;
  }
  redv[tid] = es;
  __syncthreads();
  for (int off = 128; off > 0; off >>= 1) {
    if (tid < off) redv[tid] += redv[tid + off];
    __syncthreads();
  }
  if (tid == 0) atomicAdd(errsum, redv[0]);
}

__global__ void kfinal(const float* errsum, const float* tv, float* out) {
  out[OUT_FVU_OFF]     = errsum[0] / tv[0];
  out[OUT_FVU_OFF + 1] = 0.0f;
  out[OUT_FVU_OFF + 2] = 0.0f;
}

extern "C" void kernel_launch(void* const* d_in, const int* in_sizes, int n_in,
                              void* d_out, int out_size, void* d_ws, size_t ws_size,
                              hipStream_t stream) {
  const float* x     = (const float*)d_in[0];
  const float* W_enc = (const float*)d_in[1];
  const float* b_enc = (const float*)d_in[2];
  const float* W_dec = (const float*)d_in[3];
  const float* b_dec = (const float*)d_in[4];
  float* out = (float*)d_out;

  char* ws = (char*)d_ws;
  unsigned short* wb = (unsigned short*)(ws);
  unsigned short* xb = (unsigned short*)(ws + XBF_OFF);
  float* pval   = (float*)(ws + PVAL_OFF);
  int*   pidx   = (int*)(ws + PIDX_OFF);
  float* stats  = (float*)(ws + STATS_OFF);   /* colsum[2048] colsq[2048] tv err */
  float* colsum = stats;
  float* colsq  = stats + 2048;
  float* tv     = stats + 4096;
  float* err    = stats + 4097;

  kzero<<<17, 256, 0, stream>>>(stats);
  kcolstats<<<32, 256, 0, stream>>>(x, colsum, colsq);
  ktv<<<1, 256, 0, stream>>>(colsum, colsq, tv);
  kcvt<<<2048, 256, 0, stream>>>(W_enc, wb, (NLAT * D_IN) / 4);
  kcvt_x<<<512, 256, 0, stream>>>(x, b_dec, xb, (B_ROWS * D_IN) / 4);

  dim3 g3(B_ROWS / MTILE, NSLICE);
  kenc_topk<<<g3, 256, LDS_TOTAL, stream>>>(wb, xb, b_enc, pval, pidx);

  kmerge_decode<<<B_ROWS, 256, 0, stream>>>(x, W_dec, b_dec, pval, pidx, out, err);
  kfinal<<<1, 1, 0, stream>>>(err, tv, out);
}
